// ConvDistanceTransform_45243185496346
// MI455X (gfx1250) — matrix operand order repaired and checked by reading, not compiled, not run on hardware
//
#include <hip/hip_runtime.h>
#include <hip/hip_bf16.h>

// CDNA5 (gfx1250) convolutional distance transform.
// 86 dependent iterations; each = 7x7 conv (96 x 256x256, single channel)
// mapped to 7 v_wmma_f32_16x16x32_f16 per 16x16 output tile via a banded
// Toeplitz A-matrix. Boundary kept as f16 in a 3-halo padded ping/pong pair
// in d_ws (fits in L2 together with out).

typedef __attribute__((ext_vector_type(16))) _Float16 v16h;
typedef __attribute__((ext_vector_type(8)))  float    v8f;

#define HH 256
#define WW 256
#define NIMG 96            // B*C = 32*3
#define PADH 262           // 3 + 256 + 3 rows
#define PADW 272           // 3 + 256 + 3 = 262 cols, padded to 272 halfs (row = 544B, 16B aligned)
#define ROWQ 34            // uint4 per padded row (544/16)
#define IMGQ (PADH * ROWQ) // uint4 per padded image
#define PAD_ELEMS (NIMG * PADH * PADW)   // halfs per buffer = 6,841,344
#define LDS_ROW_Q 5        // uint4 per LDS tile row (80B; 64B used; stride 80B avoids bank clustering)
#define LDS_TILE_Q (22 * LDS_ROW_Q)      // 22 halo rows per tile
#define NITER 86           // ceil(256 / (7//2))

// ---------------------------------------------------------------------------
// Init: zero d_out, zero both padded boundary buffers, fill ping interior
// from the f32 image (values are exactly 0.0 / 1.0).
// ---------------------------------------------------------------------------
__global__ __launch_bounds__(256) void cdt_init(const float* __restrict__ img,
                                                float* __restrict__ out,
                                                _Float16* __restrict__ buf0,
                                                _Float16* __restrict__ buf1) {
  int idx = blockIdx.x * 256 + threadIdx.x;
  if (idx < PAD_ELEMS) {
    int im = idx / (PADH * PADW);
    int r  = (idx % (PADH * PADW)) / PADW;
    int c  = idx % PADW;
    _Float16 v = (_Float16)0.0f;
    if (r >= 3 && r < 3 + HH && c >= 3 && c < 3 + WW)
      v = (_Float16)img[im * (HH * WW) + (r - 3) * WW + (c - 3)];
    buf0[idx] = v;
    buf1[idx] = (_Float16)0.0f;
  }
  if (idx < NIMG * HH * WW) out[idx] = 0.0f;
}

// ---------------------------------------------------------------------------
// One distance-transform iteration. One wave32 per 16x16 output tile.
//   D[m,n] = sum_dy sum_k A_dy[m,k] * B_dy[k,n]
//   A_dy[m,k] = Krow[dy][k-m]   (banded Toeplitz, shared by every tile;
//                                symmetric: A_dy == A_{6-dy} -> 4 live A's)
//   B_dy[k,n] = boundary[y0+n+dy-3][x0+k-3]   (staged through LDS)
// m = output x offset, n = output y offset, k = halo x (22 of 32 used).
// ---------------------------------------------------------------------------
__global__ __launch_bounds__(256) void cdt_step(const _Float16* __restrict__ bin,
                                                _Float16* __restrict__ bout,
                                                float* __restrict__ out,
                                                float offset) {
  __shared__ uint4 lds[8 * LDS_TILE_Q];

  const int wave = threadIdx.x >> 5;
  const int lane = threadIdx.x & 31;
  const int tile = blockIdx.x * 8 + wave;          // 3072 blocks * 8 = 24576 tiles
  const int im   = tile >> 8;                      // 256 tiles per image
  const int t    = tile & 255;
  const int y0   = (t >> 4) << 4;
  const int x0   = (t & 15) << 4;

  // ---- Stage 22x32 f16 halo tile (padded coords start exactly at (y0,x0)) ----
  const uint4* src = (const uint4*)bin;
  const int srcBase = im * IMGQ + y0 * ROWQ + (x0 >> 3);
  uint4* wl = &lds[wave * LDS_TILE_Q];
  for (int c = lane; c < 88; c += 32) {            // 22 rows * 4 x 16B chunks
    const int row = c >> 2, col = c & 3;
    wl[row * LDS_ROW_Q + col] = src[srcBase + row * ROWQ + col];
  }
  __syncthreads();                                  // LDS visibility across lanes

  // ---- Build banded Toeplitz A matrices (dy = 0..3; reuse for 6..4) ----
  const int m = lane & 15;        // A rows / D columns-in-M
  const int g = lane >> 4;
  v16h A[4];
#pragma unroll
  for (int dy = 0; dy < 4; ++dy) {
#pragma unroll
    for (int j = 0; j < 16; ++j) {
      // A layout (matches B): VGPR j <-> K = g*16 + j
      const int K  = g * 16 + j;
      const int dx = K - m;
      float val = 0.0f;
      if (dx >= 0 && dx <= 6) {
        const float fx = (float)(dx - 3), fy = (float)(dy - 3);
        val = expf(-sqrtf(fx * fx + fy * fy) * (1.0f / 0.35f));
      }
      A[dy][j] = (_Float16)val;
    }
  }

  // ---- 7 WMMAs accumulate the full 7x7 conv for the tile ----
  v8f acc = {};
  union { v16h v; uint4 q[2]; } B;
  const int n = m;                // D N-index = lane%16 = output y offset
#pragma unroll
  for (int dy = 0; dy < 7; ++dy) {
    const uint4* p = &wl[(n + dy) * LDS_ROW_Q + g * 2];   // B: VGPR j <-> K = g*16 + j
    B.q[0] = p[0];
    B.q[1] = p[1];
    const int ai = (dy < 4) ? dy : (6 - dy);
    acc = __builtin_amdgcn_wmma_f32_16x16x32_f16(
        /*neg_a=*/false, A[ai], /*neg_b=*/false, B.v,
        /*c_mod=*/(short)0, acc, /*reuse_a=*/false, /*reuse_b=*/false);
  }

  // ---- Epilogue: D VGPR v <-> (M = g*8+v, N = n) = pixel (y0+n, x0+g*8+v) ----
  // Facts (exact since center tap == 1.0 and off-center taps sum to ~0.33 < 1):
  //   * each pixel receives at most one out-contribution ever -> plain store
  //   * boundary_new = (conv > 0)
  const int y     = y0 + n;
  const int xbase = x0 + g * 8;
  float* outRow = out + im * (HH * WW) + y * WW + xbase;
  _Float16* bo  = bout + im * (PADH * PADW) + (y + 3) * PADW + (xbase + 3);
#pragma unroll
  for (int v = 0; v < 8; ++v) {
    const float conv = acc[v];
    _Float16 bnew = (_Float16)0.0f;
    if (conv > 0.0f) {
      bnew = (_Float16)1.0f;
      const float cdt = -0.35f * logf(conv);
      if (cdt > 0.0f) outRow[v] = offset + cdt;   // first & only contribution
    }
    bo[v] = bnew;
  }
}

// ---------------------------------------------------------------------------
extern "C" void kernel_launch(void* const* d_in, const int* in_sizes, int n_in,
                              void* d_out, int out_size, void* d_ws, size_t ws_size,
                              hipStream_t stream) {
  (void)in_sizes; (void)n_in; (void)out_size; (void)ws_size;
  const float* image = (const float*)d_in[0];
  float* out = (float*)d_out;

  _Float16* buf0 = (_Float16*)d_ws;            // ping (padded, f16)
  _Float16* buf1 = buf0 + PAD_ELEMS;           // pong   (2 * ~13.7 MB in d_ws)

  cdt_init<<<(PAD_ELEMS + 255) / 256, 256, 0, stream>>>(image, out, buf0, buf1);

  for (int i = 0; i < NITER; ++i) {
    const _Float16* bi = (i & 1) ? buf1 : buf0;
    _Float16*       bo = (i & 1) ? buf0 : buf1;
    cdt_step<<<(NIMG * 256) / 8, 256, 0, stream>>>(bi, bo, out, (float)i * 3.5f);
  }
}
